// Attention_34385508172472
// MI455X (gfx1250) — compile-verified, matrix operand order checked by reading, not run on hardware
//
#include <hip/hip_runtime.h>
#include <hip/hip_bf16.h>
#include <math.h>

typedef __attribute__((ext_vector_type(16))) __bf16 v16bf;
typedef __attribute__((ext_vector_type(8)))  float  v8f;
typedef int      b128vec __attribute__((vector_size(16)));  // builtin's int4 pointee
typedef unsigned u32x4   __attribute__((vector_size(16)));
typedef int      i32x4   __attribute__((vector_size(16)));
typedef int      i32x8   __attribute__((vector_size(32)));

#define B_      64
#define L_      1024
#define E_      512
#define H_      512
#define KC      32
#define NKC     (E_/KC)          // 16 k-chunks
#define BROWS   64               // rows of A per block
#define THREADS 512              // 16 waves
#define NTW     8                // N-tiles per wave (8*16 = 128 cols per wave column)

// -------- workspace layout (bytes) --------
#define WS_DEC  (512u << 10)     // dec_dense f32 [B][H]
#define WS_LOG  (640u << 10)     // logits f32 [B*L]

// ---- CDNA5 async global->LDS copy (ASYNCcnt path), guarded ----
#if defined(__gfx1250__) && __has_builtin(__builtin_amdgcn_global_load_async_to_lds_b128)
#define HAVE_ASYNC_LDS 1
#else
#define HAVE_ASYNC_LDS 0
#endif

#if defined(__gfx1250__) && __has_builtin(__builtin_amdgcn_s_wait_asynccnt)
#define WAIT_ASYNC0() __builtin_amdgcn_s_wait_asynccnt(0)
#elif defined(__gfx1250__)
#define WAIT_ASYNC0() asm volatile("s_wait_asynccnt 0x0" ::: "memory")
#else
#define WAIT_ASYNC0()
#endif

// ---- CDNA5 Tensor Data Mover (TENSORcnt path), guarded ----
#if defined(__gfx1250__) && __has_builtin(__builtin_amdgcn_tensor_load_to_lds)
#define HAVE_TDM 1
#if __has_include(<hip/amd_detail/amd_gfx1250_TDM.h>)
#define TDM_6ARG 1               // therock-10.0 headers -> 6-arg builtin
#else
#define TDM_6ARG 0               // ROCm 7.2 -> 5-arg builtin
#endif
#else
#define HAVE_TDM 0
#endif

#if defined(__gfx1250__) && __has_builtin(__builtin_amdgcn_s_wait_tensorcnt)
#define WAIT_TENSOR0() __builtin_amdgcn_s_wait_tensorcnt(0)
#elif defined(__gfx1250__)
#define WAIT_TENSOR0() asm volatile("s_wait_tensorcnt 0x0" ::: "memory")
#else
#define WAIT_TENSOR0()
#endif

#define AS_GLOBAL_B128(p) ((__attribute__((address_space(1))) b128vec*)(p))
#define AS_LDS_B128(p)    ((__attribute__((address_space(3))) b128vec*)(p))

#if HAVE_TDM
// One TDM descriptor: load a 32x64-element f32 tile (rows of 32 dwords,
// row stride 512 elements) into LDS, padding +8 dwords after every 32 dwords
// so the LDS image matches float[64][40].
//   D# group0: count=1(valid), lds_addr, 57-bit global addr, type=2
//   D# group1: data_size=4B(code 2), pad_enable, pad_interval=32dw(code 4),
//              pad_amount=8dw(code 7), tensor_dim0=512, tensor_dim1=65536,
//              tile_dim0=32, tile_dim1=64, tensor_dim0_stride=512
__device__ __forceinline__ void tdmLoadChunk(const float* gsrc, float* ldst) {
    unsigned long long ga = (unsigned long long)gsrc;
    unsigned lds_off = (unsigned)(unsigned long long)
        (__attribute__((address_space(3))) char*)ldst;
    u32x4 g0 = { 1u,                                   // count=1, user mode
                 lds_off,                              // lds_addr [63:32]
                 (unsigned)ga,                         // global_addr lo
                 (unsigned)((ga >> 32) & 0x01FFFFFFull) | 0x80000000u }; // hi + type=2
    i32x8 g1 = { (2 << 16) | (1 << 20) | (4 << 22) | (7 << 25), // data_size=4B, pad
                 (int)(512u << 16),                    // tensor_dim0[15:0] -> [63:48]
                 0,                                    // tensor_dim0 hi / dim1 lo
                 1 | (32 << 16),                       // tensor_dim1 hi=1, tile_dim0=32
                 64,                                   // tile_dim1=64, tile_dim2=0
                 512,                                  // tensor_dim0_stride lo
                 0, 0 };
    i32x4 z4 = {0, 0, 0, 0};
#if TDM_6ARG
    i32x8 z8 = {0, 0, 0, 0, 0, 0, 0, 0};
    __builtin_amdgcn_tensor_load_to_lds(g0, g1, z4, z4, z8, 0);
#else
    __builtin_amdgcn_tensor_load_to_lds(g0, g1, z4, z4, 0);
#endif
}
#endif

// ---------------------------------------------------------------------------
// Pack W_in[k][n] (f32, row-major) into bf16 fragments matching the CDNA5
// 16-bit B-matrix (32x16) VGPR layout.
// ---------------------------------------------------------------------------
__global__ __launch_bounds__(256)
void swizzleB(const float* __restrict__ W_in, __bf16* __restrict__ Bsw) {
    int idx  = blockIdx.x * 256 + threadIdx.x;   // 262144 total
    int e    = idx & 15;
    int lane = (idx >> 4) & 31;
    int kc   = (idx >> 9) & 15;
    int nt   = idx >> 13;
    int v    = e >> 1, ee = e & 1;
    int n    = nt * 16 + (lane & 15);
    int k    = kc * KC + ((v >> 2) << 4) + ((lane >> 4) << 3) + ((v & 3) << 1) + ee;
    Bsw[idx] = (__bf16)W_in[k * H_ + n];
}

__global__ __launch_bounds__(512)
void decDense(const float* __restrict__ dec, const float* __restrict__ W_dec,
              const float* __restrict__ b_dec, float* __restrict__ out) {
    int b = blockIdx.x, h = threadIdx.x;
    float s = b_dec[h];
#pragma unroll 8
    for (int k = 0; k < H_; ++k)
        s += dec[b * H_ + k] * W_dec[k * H_ + h];
    out[b * H_ + h] = s;
}

__global__ __launch_bounds__(256)
void zeroCtx(float* __restrict__ ctx) {
    int i = blockIdx.x * 256 + threadIdx.x;
    if (i < B_ * H_) ctx[i] = 0.f;
}

// ---------------------------------------------------------------------------
// Fused GEMM, double-buffered: TDM loads chunk 0, the per-wave async engine
// streams chunk kc+1 while WMMAs consume chunk kc.
// PASS2 == 0 : logits[b,l] = sum_h tanh(gemm+b_in+dec)*w_score + b_score
// PASS2 == 1 : ctx[b,h]   += sum_l weights[b,l]*(gemm+b_in)
// ---------------------------------------------------------------------------
template <int PASS2>
__global__ __launch_bounds__(THREADS)
void attnGemm(const float* __restrict__ A,        // [B*L, E] f32
              const __bf16* __restrict__ Bsw,     // swizzled W_in
              const float* __restrict__ b_in,     // [H]
              const float* __restrict__ dec_dense,// [B, H]
              const float* __restrict__ w_score,  // [H]
              const float* __restrict__ b_score,  // [1]
              const float* __restrict__ weights,  // [B*L] (pass 2)
              float* __restrict__ logits,         // [B*L] (pass 1)
              float* __restrict__ ctx)            // [B, H] (pass 2, atomic)
{
    __shared__ float Atile[2][BROWS][40];  // double-buffered, padded staging
    __shared__ float sDec[H_];
    __shared__ float sWsc[H_];
    __shared__ float sBin[H_];
    __shared__ float sLogit[BROWS];
    __shared__ float sCtx[H_];
    __shared__ float sWrow[BROWS];

    const int tid   = threadIdx.x;
    const int wave  = tid >> 5;
    const int lane  = tid & 31;
    const int mt    = wave & 3;
    const int ncol  = wave >> 2;
    const int m0    = blockIdx.x * BROWS;
    const int b     = m0 >> 10;
    const int mlane = lane & 15;
    const int hi    = lane >> 4;

    sBin[tid] = b_in[tid];
    if (PASS2) {
        sCtx[tid] = 0.f;
        if (tid < BROWS) sWrow[tid] = weights[m0 + tid];
    } else {
        sDec[tid] = dec_dense[b * H_ + tid];
        sWsc[tid] = w_score[tid];
        if (tid < BROWS) sLogit[tid] = 0.f;
    }

    const float zero = 0.f;
    v8f acc[NTW];
#pragma unroll
    for (int t = 0; t < NTW; ++t) {
        v8f z = {zero, zero, zero, zero, zero, zero, zero, zero};
        acc[t] = z;
    }

    const int arow  = tid >> 3;          // 0..63
    const int aquad = tid & 7;           // 0..7 (16B within 32-f32 chunk)

    // ---- prologue: stage chunk 0 into buffer 0 ----
#if HAVE_TDM
    if (wave == 0) {
        tdmLoadChunk(A + (size_t)m0 * E_, &Atile[0][0][0]);  // whole 64x32 tile
        WAIT_TENSOR0();
    }
#else
    {
        const float* src = A + (size_t)(m0 + arow) * E_ + aquad * 4;
#if HAVE_ASYNC_LDS
        __builtin_amdgcn_global_load_async_to_lds_b128(
            AS_GLOBAL_B128(src), AS_LDS_B128(&Atile[0][arow][aquad * 4]), 0, 0);
        WAIT_ASYNC0();
#else
        const float4 val = *(const float4*)src;
        float* dst = &Atile[0][arow][aquad * 4];
        dst[0] = val.x; dst[1] = val.y; dst[2] = val.z; dst[3] = val.w;
#endif
    }
#endif
    __syncthreads();

    for (int kc = 0; kc < NKC; ++kc) {
        const int cur = kc & 1;
        const int nxt = cur ^ 1;

        // ---- kick off async copy of the NEXT chunk into the other buffer ----
        if (kc + 1 < NKC) {
            const float* src = A + (size_t)(m0 + arow) * E_ + (kc + 1) * KC + aquad * 4;
#if HAVE_ASYNC_LDS
            __builtin_amdgcn_global_load_async_to_lds_b128(
                AS_GLOBAL_B128(src), AS_LDS_B128(&Atile[nxt][arow][aquad * 4]), 0, 0);
#else
            const float4 val = *(const float4*)src;
            float* dst = &Atile[nxt][arow][aquad * 4];
            dst[0] = val.x; dst[1] = val.y; dst[2] = val.z; dst[3] = val.w;
#endif
            if (kc + 2 < NKC)
                __builtin_prefetch(src + KC, 0, 1);
        }

        // ---- build bf16 A fragment per ISA 16-bit A layout ----
        v16bf afrag;
#pragma unroll
        for (int v = 0; v < 8; ++v) {
            const int k = ((v >> 2) << 4) + (hi << 3) + ((v & 3) << 1);
            const float2 p = *(const float2*)&Atile[cur][mt * 16 + mlane][k];
            afrag[2 * v]     = (__bf16)p.x;
            afrag[2 * v + 1] = (__bf16)p.y;
        }

        // ---- 8 WMMAs against pre-swizzled B fragments ----
#pragma unroll
        for (int t = 0; t < NTW; ++t) {
            const int ntile = ncol * 8 + t;
            const v16bf bfrag = *(const v16bf*)(Bsw +
                (((size_t)(ntile * NKC + kc) * 32 + lane) << 4));
            acc[t] = __builtin_amdgcn_wmma_f32_16x16x32_bf16(
                false, afrag, false, bfrag, (short)0, acc[t], false, false);
        }

        WAIT_ASYNC0();       // next-chunk copy landed in LDS
        __syncthreads();     // all waves done reading cur / writing nxt
    }

    if (!PASS2) {
        float part[8];
#pragma unroll
        for (int j = 0; j < 8; ++j) part[j] = 0.f;
#pragma unroll
        for (int t = 0; t < NTW; ++t) {
            const int n = ncol * 128 + t * 16 + mlane;
            const float dv = sDec[n], wv = sWsc[n], bv = sBin[n];
#pragma unroll
            for (int j = 0; j < 8; ++j)
                part[j] += tanhf(acc[t][j] + bv + dv) * wv;
        }
#pragma unroll
        for (int j = 0; j < 8; ++j) {
            const int row = mt * 16 + hi * 8 + j;
            atomicAdd(&sLogit[row], part[j]);      // ds_add_f32
        }
        __syncthreads();
        if (tid < BROWS) logits[m0 + tid] = sLogit[tid] + b_score[0];
    } else {
#pragma unroll
        for (int t = 0; t < NTW; ++t) {
            const int n = ncol * 128 + t * 16 + mlane;
            const float bv = sBin[n];
            float s = 0.f;
#pragma unroll
            for (int j = 0; j < 8; ++j) {
                const int row = mt * 16 + hi * 8 + j;
                s += sWrow[row] * (acc[t][j] + bv);
            }
            atomicAdd(&sCtx[n], s);                // ds_add_f32
        }
        __syncthreads();
        atomicAdd(&ctx[b * H_ + tid], sCtx[tid]);  // global_atomic_add_f32
    }
}

// softmax over L per batch
__global__ __launch_bounds__(256)
void softmaxL(const float* __restrict__ logits, float* __restrict__ weights) {
    __shared__ float red[256];
    const int b = blockIdx.x, tid = threadIdx.x;
    const float* lg = logits + b * L_;

    float m = -3.402823466e38f;
    for (int i = tid; i < L_; i += 256) m = fmaxf(m, lg[i]);
    red[tid] = m; __syncthreads();
    for (int s = 128; s > 0; s >>= 1) {
        if (tid < s) red[tid] = fmaxf(red[tid], red[tid + s]);
        __syncthreads();
    }
    const float mx = red[0]; __syncthreads();

    float sum = 0.f;
    for (int i = tid; i < L_; i += 256) sum += __expf(lg[i] - mx);
    red[tid] = sum; __syncthreads();
    for (int s = 128; s > 0; s >>= 1) {
        if (tid < s) red[tid] += red[tid + s];
        __syncthreads();
    }
    const float inv = 1.f / red[0];
    for (int i = tid; i < L_; i += 256)
        weights[b * L_ + i] = __expf(lg[i] - mx) * inv;
}

extern "C" void kernel_launch(void* const* d_in, const int* in_sizes, int n_in,
                              void* d_out, int out_size, void* d_ws, size_t ws_size,
                              hipStream_t stream) {
    const float* attn  = (const float*)d_in[0];  // [B,L,E]
    const float* dec   = (const float*)d_in[1];  // [B,H]
    const float* W_in  = (const float*)d_in[2];  // [E,H]
    const float* b_in  = (const float*)d_in[3];  // [H]
    const float* W_dec = (const float*)d_in[4];  // [H,H]
    const float* b_dec = (const float*)d_in[5];  // [H]
    const float* w_sc  = (const float*)d_in[6];  // [H,1]
    const float* b_sc  = (const float*)d_in[7];  // [1]

    char*  ws      = (char*)d_ws;
    __bf16* Bsw    = (__bf16*)ws;
    float* decd    = (float*)(ws + WS_DEC);
    float* logits  = (float*)(ws + WS_LOG);
    float* ctx     = (float*)d_out;              // [B,H]
    float* weights = (float*)d_out + B_ * H_;    // [B,L]

    swizzleB<<<1024, 256, 0, stream>>>(W_in, Bsw);
    decDense<<<B_, H_, 0, stream>>>(dec, W_dec, b_dec, decd);
    zeroCtx<<<(B_ * H_ + 255) / 256, 256, 0, stream>>>(ctx);

    const int gemmBlocks = (B_ * L_) / BROWS;    // 1024
    attnGemm<0><<<gemmBlocks, THREADS, 0, stream>>>(
        attn, Bsw, b_in, decd, w_sc, b_sc, nullptr, logits, nullptr);
    softmaxL<<<B_, 256, 0, stream>>>(logits, weights);
    attnGemm<1><<<gemmBlocks, THREADS, 0, stream>>>(
        attn, Bsw, b_in, decd, w_sc, b_sc, weights, nullptr, ctx);
}
